// ExplicitSVDBlock_85529978733222
// MI455X (gfx1250) — matrix operand order repaired and checked by reading, not compiled, not run on hardware
//
#include <hip/hip_runtime.h>
#include <hip/hip_bf16.h>
#include <cstdint>

// ---------------------------------------------------------------------------
// ExplicitSVDBlock for MI455X (gfx1250, wave32, WMMA f32_16x16x32_f16)
//
// All GEMM B-operands are weights, pre-transposed once into B^T [N][K] f16 so
// every LDS tile (A and B) is a contiguous 128-bit copy -> staged with the
// CDNA5 async path: global_load_async_to_lds_b128 (ASYNCcnt) + s_wait_asynccnt.
// Fragments then read back as ds_load_b128 (A-frag K-pattern and B^T rows are
// contiguous per lane).
//
// Pipeline:
//  0. prep: fuse per-head U*V -> WqkvT f16 [3072,1024]; cast/transpose weights
//  1. LN1(hidden) -> xn f16
//  2. qkv = xn @ Wqkv + bias            (WMMA GEMM, f16 out)
//  3. RoPE on q,k in place
//  4. flash attention (WMMA QK^T / PV, online softmax) -> ctx f16
//  5. x1 = ctx @ Wo^T + bo + hidden     (WMMA GEMM, f32 out)
//  6. LN2(x1) -> xn2 f16
//  7. t1 = xn2 @ U_wi                   (WMMA GEMM, f16)
//  8. g  = gelu(t1@Vwi[:, :4096]+b1) * (t1@Vwi[:, 4096:]+b2)   (WMMA, fused)
//  9. t2 = g @ U_wo                     (WMMA GEMM, f16)
// 10. out = t2 @ V_wo + b_wo + x1       (WMMA GEMM, f32 out)
// ---------------------------------------------------------------------------

typedef __attribute__((ext_vector_type(16))) _Float16 v16h;
typedef __attribute__((ext_vector_type(8)))  float    v8f;

#define NB   4
#define NM   2048
#define ND   1024
#define NH   16
#define NDH  64
#define NROWS (NB * NM)          // 8192 token rows

__device__ __forceinline__ v8f wmma_f16(v16h a, v16h b, v8f c) {
  return __builtin_amdgcn_wmma_f32_16x16x32_f16(
      /*neg_a=*/false, a, /*neg_b=*/false, b,
      /*c_mod=*/(short)0, c, /*reuse_a=*/false, /*reuse_b=*/false);
}

// K index inside a 16x32 f16 A-fragment for element e (0..15) of lane-half kh.
// Contiguous order matching the B reads: K = 16*kh + e.
// Elements 0..15 are one contiguous 16-half run -> 2x ds_load_b128.
__device__ __forceinline__ int a_frag_k(int e, int kh) {
  return kh * 16 + e;
}

__device__ __forceinline__ float gelu_tanh(float x) {
  float x3 = x * x * x;
  return 0.5f * x * (1.0f + tanhf(0.7978845608028654f * (x + 0.044715f * x3)));
}

// CDNA5 async global->LDS copy, 16B per lane (ISA 08_async_tensor 4.2/4.4, GV
// mode): copies from gaddr into LDS at byte offset lds_addr. Tracked by ASYNCcnt.
__device__ __forceinline__ void async_copy_b128(unsigned lds_addr,
                                                uint64_t gaddr) {
  asm volatile("global_load_async_to_lds_b128 %0, %1, off"
               :: "v"(lds_addr), "v"(gaddr) : "memory");
}
__device__ __forceinline__ void wait_async0() {
  asm volatile("s_wait_asynccnt 0" ::: "memory");
}
// Generic->LDS offset: low 32 bits of the flat address (LDS aperture keeps the
// byte offset in addr[31:0]).
__device__ __forceinline__ unsigned lds_off(const void* p) {
  return (unsigned)(uintptr_t)p;
}

// ---------------------------------------------------------------------------
// WMMA GEMM: C[M,N] = A[M,K] @ Bt^T (+bias) (+resid); Bt is B^T in [N][K].
// Block: 256 threads (8 waves). Macro-tile 128x64, K-step 32, double-buffered
// async staging. Wave w owns rows w*16..+15, all 64 cols (4 WMMA / K-step).
// ---------------------------------------------------------------------------
template <bool BIAS, bool RES, bool OUTF16>
__global__ __launch_bounds__(256) void gemm_wmma(
    const _Float16* __restrict__ A, int lda,
    const _Float16* __restrict__ Bt, int ldb,   // ldb = K stride of B^T rows
    const float* __restrict__ bias, const float* __restrict__ resid,
    _Float16* __restrict__ Ch, float* __restrict__ Cf, int ldc, int K) {
  __shared__ __align__(16) _Float16 sA[2][128][40];   // [buf][m][k]
  __shared__ __align__(16) _Float16 sBt[2][64][40];   // [buf][n][k]

  const int tid  = threadIdx.x;
  const int wave = tid >> 5;
  const int lane = tid & 31;
  const int row  = lane & 15;
  const int kh   = lane >> 4;
  const size_t m0 = (size_t)blockIdx.y * 128;
  const int    n0 = blockIdx.x * 64;
  const int    wm = wave * 16;

  v8f acc[4] = {};

  const int ar = tid >> 1, ac = (tid & 1) * 16;  // A: row ar, halves ac..ac+15
  const int br = tid >> 2, bc = (tid & 3) * 8;   // B^T: row n=br, halves bc..+7

  auto stage = [&](int k0, int buf) {
    unsigned la = lds_off(&sA[buf][ar][ac]);
    uint64_t ga = (uint64_t)(uintptr_t)(A + (m0 + ar) * lda + k0 + ac);
    async_copy_b128(la, ga);
    async_copy_b128(la + 16, ga + 16);
    unsigned lb = lds_off(&sBt[buf][br][bc]);
    uint64_t gb = (uint64_t)(uintptr_t)(Bt + (size_t)(n0 + br) * ldb + k0 + bc);
    async_copy_b128(lb, gb);
  };

  stage(0, 0);
  int cur = 0;
  for (int k0 = 0; k0 < K; k0 += 32) {
    wait_async0();
    __syncthreads();
    if (k0 + 32 < K) stage(k0 + 32, cur ^ 1);
    v16h af, bf0, bf1, bf2, bf3;
#pragma unroll
    for (int e = 0; e < 16; ++e) af[e] = sA[cur][wm + row][a_frag_k(e, kh)];
#pragma unroll
    for (int e = 0; e < 16; ++e) {
      int kk = kh * 16 + e;
      bf0[e] = sBt[cur][row][kk];
      bf1[e] = sBt[cur][16 + row][kk];
      bf2[e] = sBt[cur][32 + row][kk];
      bf3[e] = sBt[cur][48 + row][kk];
    }
    acc[0] = wmma_f16(af, bf0, acc[0]);
    acc[1] = wmma_f16(af, bf1, acc[1]);
    acc[2] = wmma_f16(af, bf2, acc[2]);
    acc[3] = wmma_f16(af, bf3, acc[3]);
    cur ^= 1;
  }

#pragma unroll
  for (int j = 0; j < 4; ++j) {
#pragma unroll
    for (int i = 0; i < 8; ++i) {
      size_t gm = m0 + wm + kh * 8 + i;  // C layout: lanes 0-15 rows 0-7, 16-31 rows 8-15
      int gn = n0 + j * 16 + row;
      float v = acc[j][i];
      if (BIAS) v += bias[gn];
      if (RES)  v += resid[gm * ldc + gn];
      if (OUTF16) Ch[gm * ldc + gn] = (_Float16)v;
      else        Cf[gm * ldc + gn] = v;
    }
  }
}

// ---------------------------------------------------------------------------
// GEGLU GEMM: g[M,4096] = gelu(A@Vwi[:, :4096]+b1) * (A@Vwi[:, 4096:]+b2)
// VwiT is V_wi^T [8192][512]; the two halves are row ranges n and 4096+n.
// ---------------------------------------------------------------------------
__global__ __launch_bounds__(256) void gemm_geglu(
    const _Float16* __restrict__ A, int lda,
    const _Float16* __restrict__ Bt, int ldb,   // [8192][512], ldb=512
    const float* __restrict__ bias,             // length 8192
    _Float16* __restrict__ G, int ldg, int K) {
  __shared__ __align__(16) _Float16 sA[2][128][40];
  __shared__ __align__(16) _Float16 sB1t[2][64][40];
  __shared__ __align__(16) _Float16 sB2t[2][64][40];

  const int tid  = threadIdx.x;
  const int wave = tid >> 5;
  const int lane = tid & 31;
  const int row  = lane & 15;
  const int kh   = lane >> 4;
  const size_t m0 = (size_t)blockIdx.y * 128;
  const int    n0 = blockIdx.x * 64;
  const int    wm = wave * 16;

  v8f acc1[4] = {}, acc2[4] = {};

  const int ar = tid >> 1, ac = (tid & 1) * 16;
  const int br = tid >> 2, bc = (tid & 3) * 8;

  auto stage = [&](int k0, int buf) {
    unsigned la = lds_off(&sA[buf][ar][ac]);
    uint64_t ga = (uint64_t)(uintptr_t)(A + (m0 + ar) * lda + k0 + ac);
    async_copy_b128(la, ga);
    async_copy_b128(la + 16, ga + 16);
    unsigned lb1 = lds_off(&sB1t[buf][br][bc]);
    uint64_t gb1 = (uint64_t)(uintptr_t)(Bt + (size_t)(n0 + br) * ldb + k0 + bc);
    async_copy_b128(lb1, gb1);
    unsigned lb2 = lds_off(&sB2t[buf][br][bc]);
    uint64_t gb2 =
        (uint64_t)(uintptr_t)(Bt + (size_t)(4096 + n0 + br) * ldb + k0 + bc);
    async_copy_b128(lb2, gb2);
  };

  stage(0, 0);
  int cur = 0;
  for (int k0 = 0; k0 < K; k0 += 32) {
    wait_async0();
    __syncthreads();
    if (k0 + 32 < K) stage(k0 + 32, cur ^ 1);
    v16h af;
#pragma unroll
    for (int e = 0; e < 16; ++e) af[e] = sA[cur][wm + row][a_frag_k(e, kh)];
#pragma unroll
    for (int j = 0; j < 4; ++j) {
      v16h b1f, b2f;
#pragma unroll
      for (int e = 0; e < 16; ++e) {
        int kk = kh * 16 + e;
        b1f[e] = sB1t[cur][j * 16 + row][kk];
        b2f[e] = sB2t[cur][j * 16 + row][kk];
      }
      acc1[j] = wmma_f16(af, b1f, acc1[j]);
      acc2[j] = wmma_f16(af, b2f, acc2[j]);
    }
    cur ^= 1;
  }

#pragma unroll
  for (int j = 0; j < 4; ++j) {
#pragma unroll
    for (int i = 0; i < 8; ++i) {
      size_t gm = m0 + wm + kh * 8 + i;
      int gn = n0 + j * 16 + row;
      float h1 = acc1[j][i] + bias[gn];
      float h2 = acc2[j][i] + bias[4096 + gn];
      G[gm * ldg + gn] = (_Float16)(gelu_tanh(h1) * h2);
    }
  }
}

// ---------------------------------------------------------------------------
// Flash attention: grid (M/64, H, B), block 128 (4 waves).
// Each wave owns a 16-query tile; K/V staged in LDS in 32-key blocks.
// qkv layout: [row 0..8191][3072] f16, cols: q h*64 | 1024+k h*64 | 2048+v h*64
// sK row-major (QK^T B-frag reads k-contiguous, staged async); sV transposed
// [dh][key] so PV B-frag reads are key-contiguous.
// ---------------------------------------------------------------------------
__global__ __launch_bounds__(128) void attn_flash(
    const _Float16* __restrict__ qkv, const int* __restrict__ mask,
    _Float16* __restrict__ ctx) {
  __shared__ __align__(16) _Float16 sK[32][72];    // [key][dh]
  __shared__ __align__(16) _Float16 sVt[64][40];   // [dh][key]
  __shared__ __align__(16) _Float16 sP[4][16][40]; // [wave][q][key]

  const int tid  = threadIdx.x;
  const int wave = tid >> 5;
  const int lane = tid & 31;
  const int row  = lane & 15;
  const int half = lane >> 4;
  const int h = blockIdx.y, b = blockIdx.z;
  const int q0 = blockIdx.x * 64 + wave * 16;
  const size_t rowbase = (size_t)b * NM;
  const int qcol = h * 64;
  const int kcol = 1024 + h * 64;
  const int vcol = 2048 + h * 64;

  // Q fragments for K-chunks dh 0..31 / 32..63, pre-scaled by 1/sqrt(dh)
  v16h qf0, qf1;
#pragma unroll
  for (int e = 0; e < 16; ++e) {
    int kk = a_frag_k(e, half);
    const _Float16* qp = qkv + (rowbase + q0 + row) * 3072 + qcol;
    qf0[e] = (_Float16)((float)qp[kk] * 0.125f);
    qf1[e] = (_Float16)((float)qp[32 + kk] * 0.125f);
  }

  float mrow[8], lrow[8];
  v8f o0 = {}, o1 = {}, o2 = {}, o3 = {};
#pragma unroll
  for (int i = 0; i < 8; ++i) { mrow[i] = -3.0e38f; lrow[i] = 0.0f; }

  const int r = tid >> 2, c = (tid & 3) * 16;

  for (int kb = 0; kb < NM; kb += 32) {
    // K tile: straight async copy (key-row-major)
    {
      unsigned lk = lds_off(&sK[r][c]);
      uint64_t gk =
          (uint64_t)(uintptr_t)(qkv + (rowbase + kb + r) * 3072 + kcol + c);
      async_copy_b128(lk, gk);
      async_copy_b128(lk + 16, gk + 16);
      // V tile: transpose while staging
      const float4* vs =
          (const float4*)(qkv + (rowbase + kb + r) * 3072 + vcol + c);
      float4 v0 = vs[0], v1 = vs[1];
      const _Float16* vh0 = (const _Float16*)&v0;
      const _Float16* vh1 = (const _Float16*)&v1;
#pragma unroll
      for (int j = 0; j < 8; ++j) {
        sVt[c + j][r]     = vh0[j];
        sVt[c + 8 + j][r] = vh1[j];
      }
    }
    wait_async0();
    __syncthreads();

    // S = Q @ K^T : two 16x16 key tiles, K-dim 64 = 2 WMMA each
    v8f s0 = {}, s1 = {};
    {
      v16h kt;
#pragma unroll
      for (int e = 0; e < 16; ++e) kt[e] = sK[row][half * 16 + e];
      s0 = wmma_f16(qf0, kt, s0);
#pragma unroll
      for (int e = 0; e < 16; ++e) kt[e] = sK[row][32 + half * 16 + e];
      s0 = wmma_f16(qf1, kt, s0);
#pragma unroll
      for (int e = 0; e < 16; ++e) kt[e] = sK[16 + row][half * 16 + e];
      s1 = wmma_f16(qf0, kt, s1);
#pragma unroll
      for (int e = 0; e < 16; ++e) kt[e] = sK[16 + row][32 + half * 16 + e];
      s1 = wmma_f16(qf1, kt, s1);
    }

    const int mk0 = mask[(size_t)b * NM + kb + row];
    const int mk1 = mask[(size_t)b * NM + kb + 16 + row];

    // online softmax; row i lives across the 16 lanes of this half-group
#pragma unroll
    for (int i = 0; i < 8; ++i) {
      float v0 = mk0 ? s0[i] : -3.0e38f;
      float v1 = mk1 ? s1[i] : -3.0e38f;
      float mx = fmaxf(v0, v1);
      mx = fmaxf(mx, __shfl_xor(mx, 1));
      mx = fmaxf(mx, __shfl_xor(mx, 2));
      mx = fmaxf(mx, __shfl_xor(mx, 4));
      mx = fmaxf(mx, __shfl_xor(mx, 8));
      float nm = fmaxf(mrow[i], mx);
      float corr = __expf(mrow[i] - nm);
      mrow[i] = nm;
      float p0 = __expf(v0 - nm);
      float p1 = __expf(v1 - nm);
      float ps = p0 + p1;
      ps += __shfl_xor(ps, 1);
      ps += __shfl_xor(ps, 2);
      ps += __shfl_xor(ps, 4);
      ps += __shfl_xor(ps, 8);
      lrow[i] = lrow[i] * corr + ps;
      o0[i] *= corr; o1[i] *= corr; o2[i] *= corr; o3[i] *= corr;
      int rr = half * 8 + i;
      sP[wave][rr][row]      = (_Float16)p0;
      sP[wave][rr][16 + row] = (_Float16)p1;
    }
    __syncthreads();

    // O += P @ V  (A = P 16x32; B = V as four 16-dh-col tiles, key-contiguous)
    v16h pf, vf0, vf1, vf2, vf3;
#pragma unroll
    for (int e = 0; e < 16; ++e) pf[e] = sP[wave][row][a_frag_k(e, half)];
#pragma unroll
    for (int e = 0; e < 16; ++e) {
      int kk = half * 16 + e;
      vf0[e] = sVt[row][kk];
      vf1[e] = sVt[16 + row][kk];
      vf2[e] = sVt[32 + row][kk];
      vf3[e] = sVt[48 + row][kk];
    }
    o0 = wmma_f16(pf, vf0, o0);
    o1 = wmma_f16(pf, vf1, o1);
    o2 = wmma_f16(pf, vf2, o2);
    o3 = wmma_f16(pf, vf3, o3);
    __syncthreads();
  }

#pragma unroll
  for (int i = 0; i < 8; ++i) {
    float inv = 1.0f / lrow[i];
    size_t gr = rowbase + q0 + half * 8 + i;
    _Float16* dst = ctx + gr * 1024 + h * 64;
    dst[row]      = (_Float16)(o0[i] * inv);
    dst[16 + row] = (_Float16)(o1[i] * inv);
    dst[32 + row] = (_Float16)(o2[i] * inv);
    dst[48 + row] = (_Float16)(o3[i] * inv);
  }
}

// ---------------------------------------------------------------------------
// LayerNorm (row of 1024) -> f16. One block (256 threads) per row.
// ---------------------------------------------------------------------------
__global__ __launch_bounds__(256) void ln_kernel(
    const float* __restrict__ x, const float* __restrict__ w,
    const float* __restrict__ bsc, _Float16* __restrict__ out) {
  __shared__ float rs[8], rss[8];
  const size_t r = blockIdx.x;
  const float* xr = x + r * ND;
  float vals[4], s = 0.f, ss = 0.f;
#pragma unroll
  for (int i = 0; i < 4; ++i) {
    float v = xr[threadIdx.x + i * 256];
    vals[i] = v; s += v; ss += v * v;
  }
  for (int m = 1; m < 32; m <<= 1) { s += __shfl_xor(s, m); ss += __shfl_xor(ss, m); }
  int wave = threadIdx.x >> 5, lane = threadIdx.x & 31;
  if (lane == 0) { rs[wave] = s; rss[wave] = ss; }
  __syncthreads();
  s = 0.f; ss = 0.f;
#pragma unroll
  for (int i = 0; i < 8; ++i) { s += rs[i]; ss += rss[i]; }
  float mu = s * (1.0f / ND);
  float var = ss * (1.0f / ND) - mu * mu;
  float rstd = rsqrtf(var + 1e-5f);
#pragma unroll
  for (int i = 0; i < 4; ++i) {
    int c = threadIdx.x + i * 256;
    out[r * ND + c] = (_Float16)((vals[i] - mu) * rstd * w[c] + bsc[c]);
  }
}

// ---------------------------------------------------------------------------
// RoPE in place on q,k halves of qkv. One thread per (row, head, d<32).
// ---------------------------------------------------------------------------
__global__ void rope_kernel(_Float16* __restrict__ qkv) {
  size_t idx = (size_t)blockIdx.x * blockDim.x + threadIdx.x;
  if (idx >= (size_t)NROWS * NH * 32) return;
  int d = idx & 31;
  int h = (idx >> 5) & 15;
  size_t rowi = idx >> 9;
  int m = (int)(rowi & (NM - 1));
  float freq = __powf(10000.0f, -(float)(2 * d) / 64.0f);
  float ang = (float)m * freq;
  float c, sn;
  __sincosf(ang, &sn, &c);
  _Float16* qp = qkv + rowi * 3072 + h * 64 + d;
  float q1 = (float)qp[0], q2 = (float)qp[32];
  qp[0]  = (_Float16)(q1 * c - q2 * sn);
  qp[32] = (_Float16)(q2 * c + q1 * sn);
  _Float16* kp = qkv + rowi * 3072 + 1024 + h * 64 + d;
  float k1 = (float)kp[0], k2 = (float)kp[32];
  kp[0]  = (_Float16)(k1 * c - k2 * sn);
  kp[32] = (_Float16)(k2 * c + k1 * sn);
}

// ---------------------------------------------------------------------------
// Fuse per-head low-rank U@V -> WqkvT f16 [3072][1024]  (B^T layout [n][k=d]).
// grid (8 d-tiles, 16 heads, 3 projections), block 128.
// ---------------------------------------------------------------------------
__global__ __launch_bounds__(128) void prep_wqkv(
    const float* __restrict__ Uq, const float* __restrict__ Vq,
    const float* __restrict__ Uk, const float* __restrict__ Vk,
    const float* __restrict__ Uv, const float* __restrict__ Vv,
    _Float16* __restrict__ WqkvT) {
  const int p = blockIdx.z, h = blockIdx.y, dt = blockIdx.x;
  const float* U = (p == 0 ? Uq : (p == 1 ? Uk : Uv)) + (size_t)h * ND * 32;
  const float* V = (p == 0 ? Vq : (p == 1 ? Vk : Vv)) + (size_t)h * 32 * 64;
  __shared__ float sV[32][64];
  for (int i = threadIdx.x; i < 2048; i += 128) sV[i >> 6][i & 63] = V[i];
  __syncthreads();
  const int d = dt * 128 + threadIdx.x;
  float u[32];
#pragma unroll
  for (int rr = 0; rr < 32; ++rr) u[rr] = U[(size_t)d * 32 + rr];
  for (int j = 0; j < 64; ++j) {
    float acc = 0.f;
#pragma unroll
    for (int rr = 0; rr < 32; ++rr) acc += u[rr] * sV[rr][j];
    WqkvT[(size_t)(p * 1024 + h * 64 + j) * ND + d] = (_Float16)acc;
  }
}

__global__ void pack_qkv_bias(const float* __restrict__ bq,
                              const float* __restrict__ bk,
                              const float* __restrict__ bv,
                              float* __restrict__ out) {
  int i = blockIdx.x * blockDim.x + threadIdx.x;
  if (i >= 3072) return;
  const float* src = (i < 1024) ? bq : (i < 2048 ? bk : bv);
  out[i] = src[i & 1023];
}

__global__ void cast_f16(const float* __restrict__ in, _Float16* __restrict__ out,
                         size_t n) {
  for (size_t i = (size_t)blockIdx.x * blockDim.x + threadIdx.x; i < n;
       i += (size_t)gridDim.x * blockDim.x)
    out[i] = (_Float16)in[i];
}

// out[c][r] = in[r][c], f32 -> f16 (tiled); used to build B^T weight layouts.
__global__ __launch_bounds__(256) void transpose_cast(
    const float* __restrict__ in, _Float16* __restrict__ out, int rows, int cols) {
  __shared__ float tile[32][33];
  int c0 = blockIdx.x * 32, r0 = blockIdx.y * 32;
  int tx = threadIdx.x & 31, ty = threadIdx.x >> 5;
  for (int i = 0; i < 32; i += 8)
    tile[ty + i][tx] = in[(size_t)(r0 + ty + i) * cols + c0 + tx];
  __syncthreads();
  for (int i = 0; i < 32; i += 8)
    out[(size_t)(c0 + ty + i) * rows + r0 + tx] = (_Float16)tile[tx][ty + i];
}

// ---------------------------------------------------------------------------
// Host launcher
// ---------------------------------------------------------------------------
extern "C" void kernel_launch(void* const* d_in, const int* in_sizes, int n_in,
                              void* d_out, int out_size, void* d_ws,
                              size_t ws_size, hipStream_t stream) {
  const float* hidden = (const float*)d_in[0];
  const int*   mask   = (const int*)d_in[1];
  const float* ln1w = (const float*)d_in[2];
  const float* ln1b = (const float*)d_in[3];
  const float* Uq = (const float*)d_in[4];
  const float* Vq = (const float*)d_in[5];
  const float* bq = (const float*)d_in[6];
  const float* Uk = (const float*)d_in[7];
  const float* Vk = (const float*)d_in[8];
  const float* bk = (const float*)d_in[9];
  const float* Uv = (const float*)d_in[10];
  const float* Vv = (const float*)d_in[11];
  const float* bv = (const float*)d_in[12];
  const float* Wo = (const float*)d_in[13];
  const float* bo = (const float*)d_in[14];
  const float* ln2w = (const float*)d_in[15];
  const float* ln2b = (const float*)d_in[16];
  const float* U_wi = (const float*)d_in[17];
  const float* V_wi = (const float*)d_in[18];
  const float* b_wi = (const float*)d_in[19];
  const float* U_wo = (const float*)d_in[20];
  const float* V_wo = (const float*)d_in[21];
  const float* b_wo = (const float*)d_in[22];
  float* out = (float*)d_out;

  char* ws = (char*)d_ws;
  // workspace layout (lifetime-overlapped regions)
  constexpr size_t SZ_XN  = (size_t)NROWS * ND * 2;      // 16 MB (xn / xn2)
  constexpr size_t SZ_BIG = (size_t)NROWS * 4096 * 2;    // 64 MB (qkv 48MB, then g)
  constexpr size_t SZ_CT  = (size_t)NROWS * ND * 2;      // 16 MB (ctx, then t1|t2)
  constexpr size_t SZ_X1  = (size_t)NROWS * ND * 4;      // 32 MB

  size_t off = 0;
  auto take = [&](size_t bytes) { char* p = ws + off; off += (bytes + 255) & ~(size_t)255; return p; };
  _Float16* xn    = (_Float16*)take(SZ_XN);
  char*     bigp  = take(SZ_BIG);
  _Float16* qkv   = (_Float16*)bigp;                 // phase 1
  _Float16* g     = (_Float16*)bigp;                 // phase 2 (qkv dead)
  char*     ctp   = take(SZ_CT);
  _Float16* ctx   = (_Float16*)ctp;                  // phase 1
  _Float16* t1    = (_Float16*)ctp;                  // phase 2 (ctx dead)
  _Float16* t2    = (_Float16*)(ctp + (size_t)NROWS * 512 * 2);
  float*    x1    = (float*)take(SZ_X1);
  _Float16* WqkvT = (_Float16*)take((size_t)3072 * ND * 2);  // [3072][1024]
  float*    qbias = (float*)take(3072 * 4);
  _Float16* wo16  = (_Float16*)take((size_t)ND * ND * 2);    // Wo as-is == B^T
  _Float16* uwiT  = (_Float16*)take((size_t)512 * ND * 2);   // [512][1024]
  _Float16* vwiT  = (_Float16*)take((size_t)8192 * 512 * 2); // [8192][512]
  _Float16* uwoT  = (_Float16*)take((size_t)512 * 4096 * 2); // [512][4096]
  _Float16* vwoT  = (_Float16*)take((size_t)ND * 512 * 2);   // [1024][512]

  // ---- weight prep (all B operands in B^T [N][K] layout) ----
  prep_wqkv<<<dim3(8, 16, 3), 128, 0, stream>>>(Uq, Vq, Uk, Vk, Uv, Vv, WqkvT);
  pack_qkv_bias<<<12, 256, 0, stream>>>(bq, bk, bv, qbias);
  cast_f16<<<1024, 256, 0, stream>>>(Wo, wo16, (size_t)ND * ND);  // Wo == (Wo^T)^T
  transpose_cast<<<dim3(512 / 32, ND / 32), 256, 0, stream>>>(U_wi, uwiT, ND, 512);
  transpose_cast<<<dim3(8192 / 32, 512 / 32), 256, 0, stream>>>(V_wi, vwiT, 512, 8192);
  transpose_cast<<<dim3(512 / 32, 4096 / 32), 256, 0, stream>>>(U_wo, uwoT, 4096, 512);
  transpose_cast<<<dim3(ND / 32, 512 / 32), 256, 0, stream>>>(V_wo, vwoT, 512, ND);

  // ---- 1. LN1 ----
  ln_kernel<<<NROWS, 256, 0, stream>>>(hidden, ln1w, ln1b, xn);

  // ---- 2. QKV GEMM: [8192,1024] x [1024,3072] + bias -> f16 ----
  gemm_wmma<true, false, true><<<dim3(3072 / 64, NROWS / 128), 256, 0, stream>>>(
      xn, ND, WqkvT, ND, qbias, nullptr, qkv, nullptr, 3072, ND);

  // ---- 3. RoPE ----
  rope_kernel<<<(NROWS * NH * 32 + 255) / 256, 256, 0, stream>>>(qkv);

  // ---- 4. attention ----
  attn_flash<<<dim3(NM / 64, NH, NB), 128, 0, stream>>>(qkv, mask, ctx);

  // ---- 5. Wo GEMM + residual -> x1 (f32) ----
  gemm_wmma<true, true, false><<<dim3(ND / 64, NROWS / 128), 256, 0, stream>>>(
      ctx, ND, wo16, ND, bo, hidden, nullptr, x1, ND, ND);

  // ---- 6. LN2 ----
  ln_kernel<<<NROWS, 256, 0, stream>>>(x1, ln2w, ln2b, xn);

  // ---- 7. t1 = xn2 @ U_wi ----
  gemm_wmma<false, false, true><<<dim3(512 / 64, NROWS / 128), 256, 0, stream>>>(
      xn, ND, uwiT, ND, nullptr, nullptr, t1, nullptr, 512, ND);

  // ---- 8. g = GEGLU(t1 @ V_wi + b_wi) ----
  gemm_geglu<<<dim3(4096 / 64, NROWS / 128), 256, 0, stream>>>(
      t1, 512, vwiT, 512, b_wi, g, 4096, 512);

  // ---- 9. t2 = g @ U_wo ----
  gemm_wmma<false, false, true><<<dim3(512 / 64, NROWS / 128), 256, 0, stream>>>(
      g, 4096, uwoT, 4096, nullptr, nullptr, t2, nullptr, 512, 4096);

  // ---- 10. out = t2 @ V_wo + b_wo + x1 ----
  gemm_wmma<true, true, false><<<dim3(ND / 64, NROWS / 128), 256, 0, stream>>>(
      t2, 512, vwoT, 512, b_wo, x1, nullptr, out, ND, 512);

  (void)in_sizes; (void)n_in; (void)out_size; (void)ws_size;
}
